// GNN_62861141344694
// MI455X (gfx1250) — matrix operand order repaired and checked by reading, not compiled, not run on hardware
//
#include <hip/hip_runtime.h>
#include <stdint.h>

// ---------------------------------------------------------------------------
// GraphConv (aggr='max') x4 layers for MI455X (gfx1250, wave32, WMMA + TDM).
//  - CSR built once per call (atomics only in setup, not in hot loop)
//  - bf16 activations (halves gather traffic; activations stay L2-resident)
//  - v_wmma_f32_16x16x32_bf16 with f32 accumulate; weight tiles DMA'd into
//    LDS by the Tensor Data Mover (tensor_load_to_lds / s_wait_tensorcnt),
//    inner loop is ds_load_b128 -> wmma.
// ---------------------------------------------------------------------------

typedef __attribute__((ext_vector_type(8)))  __bf16 v8bf;
typedef __attribute__((ext_vector_type(16))) __bf16 v16bf;
typedef __attribute__((ext_vector_type(8)))  float  v8f;
typedef __attribute__((ext_vector_type(4)))  unsigned int u32x4;
typedef __attribute__((ext_vector_type(4)))  int          i32x4;
typedef __attribute__((ext_vector_type(8)))  int          i32x8;

#define FEAT 128

// Toolchain arity hedge: therock headers ship the 6-arg TDM builtin,
// ROCm 7.2 (clang-22) declares the 5-arg form.
#if defined(__has_include)
#if __has_include(<hip/amd_detail/amd_gfx1250_TDM.h>)
#define TDM_6ARG 1
#else
#define TDM_6ARG 0
#endif
#else
#define TDM_6ARG 0
#endif

__device__ __forceinline__ float bfu_to_f(unsigned short u) {
  return __builtin_bit_cast(float, ((unsigned)u) << 16);
}
__device__ __forceinline__ unsigned short f_to_bfu(float f) {
  unsigned u = __builtin_bit_cast(unsigned, f);
  u += 0x7FFFu + ((u >> 16) & 1u);        // round to nearest even
  return (unsigned short)(u >> 16);
}

// TDM: DMA `bytes` contiguous bytes from global `gaddr` to LDS offset
// `lds_off`, expressed as a 1-row tile of (bytes/4) 4-byte elements.
// D# layouts per CDNA5 ISA 8.3/8.4 (group0: count/lds/global/type;
// group1: data_size, tensor_dim0/1, tile_dim0/1, dim0 stride).
__device__ __forceinline__ void tdm_load_to_lds(unsigned long long gaddr,
                                                unsigned lds_off,
                                                unsigned bytes) {
  unsigned n = bytes >> 2;               // 4-byte elements
  u32x4 g0;
  g0[0] = 1u;                                          // count=1, user mode
  g0[1] = lds_off;                                     // LDS byte address
  g0[2] = (unsigned)(gaddr & 0xFFFFFFFFull);           // global_addr[31:0]
  g0[3] = (unsigned)((gaddr >> 32) & 0x01FFFFFFull)    // global_addr[56:32]
          | (2u << 30);                                // type = 2 ("image")
  i32x8 g1;
  g1[0] = (int)(2u << 16);                             // data_size=4B, no mask
  g1[1] = (int)((n & 0xFFFFu) << 16);                  // tensor_dim0[15:0]
  g1[2] = (int)((n >> 16) | (1u << 16));               // dim0 hi | tensor_dim1=1
  g1[3] = (int)((n & 0xFFFFu) << 16);                  // tile_dim0 = n
  g1[4] = 1;                                           // tile_dim1 = 1
  g1[5] = (int)n;                                      // tensor_dim0_stride lo
  g1[6] = 0;
  g1[7] = 0;
  i32x4 gz = {0, 0, 0, 0};
#if TDM_6ARG
  i32x8 gz8 = {0, 0, 0, 0, 0, 0, 0, 0};
  __builtin_amdgcn_tensor_load_to_lds(g0, g1, gz, gz, gz8, 0);
#else
  __builtin_amdgcn_tensor_load_to_lds(g0, g1, gz, gz, 0);
#endif
}

// ----------------------------- CSR construction ----------------------------

__global__ void k_zero_i32(int* __restrict__ p, int n) {
  int i = blockIdx.x * blockDim.x + threadIdx.x;
  if (i < n) p[i] = 0;
}

__global__ void k_hist(const int* __restrict__ dst, int* __restrict__ cnt, int e) {
  for (int i = blockIdx.x * blockDim.x + threadIdx.x; i < e;
       i += gridDim.x * blockDim.x)
    atomicAdd(&cnt[dst[i]], 1);
}

// Single-block exclusive scan over node degrees -> row pointers.
// Also re-zeroes cnt[] so it can serve as the fill cursor.
__global__ void k_scan(int* __restrict__ cnt, int* __restrict__ ptr, int n) {
  __shared__ int sm[1024];
  __shared__ int s_run;
  int t = threadIdx.x;
  if (t == 0) s_run = 0;
  __syncthreads();
  for (int base = 0; base < n; base += 1024) {
    int i = base + t;
    int v = 0;
    if (i < n) { v = cnt[i]; cnt[i] = 0; }
    sm[t] = v;
    __syncthreads();
    #pragma unroll
    for (int off = 1; off < 1024; off <<= 1) {
      int x = (t >= off) ? sm[t - off] : 0;
      __syncthreads();
      sm[t] += x;
      __syncthreads();
    }
    if (i < n) ptr[i] = s_run + sm[t] - v;   // exclusive prefix
    __syncthreads();
    if (t == 0) s_run += sm[1023];
    __syncthreads();
  }
  if (t == 0) ptr[n] = s_run;
}

__global__ void k_fill(const int* __restrict__ src, const int* __restrict__ dst,
                       const int* __restrict__ ptr, int* __restrict__ cur,
                       int* __restrict__ csr, int e) {
  for (int i = blockIdx.x * blockDim.x + threadIdx.x; i < e;
       i += gridDim.x * blockDim.x) {
    int d   = dst[i];
    int pos = ptr[d] + atomicAdd(&cur[d], 1);
    csr[pos] = src[i];
  }
}

// ------------------------------- bf16 casts --------------------------------

__global__ void k_cast_bf16(const float* __restrict__ in,
                            unsigned short* __restrict__ out, int n) {
  int i = blockIdx.x * blockDim.x + threadIdx.x;
  if (i < n) out[i] = f_to_bfu(in[i]);
}

// W f32 [K][N] row-major  ->  Wt bf16 [N][K] (so WMMA B-fragments are
// contiguous 32B runs per lane).
__global__ void k_transpose_bf16(const float* __restrict__ in,
                                 unsigned short* __restrict__ out) {
  int t = blockIdx.x * blockDim.x + threadIdx.x;   // 0..16383
  int nrow = t >> 7, k = t & 127;
  out[t] = f_to_bfu(in[k * FEAT + nrow]);
}

// ----------------------- per-node gather-max aggregation -------------------
// One block per destination node, one thread per feature channel. Unrolled
// by 4 so four coalesced 256B gathers are in flight per iteration.

__global__ void k_aggregate(const unsigned short* __restrict__ h,
                            const int* __restrict__ ptr,
                            const int* __restrict__ csr,
                            unsigned short* __restrict__ agg) {
  int n = blockIdx.x;
  int c = threadIdx.x;                 // 0..127
  int b = ptr[n], e = ptr[n + 1];
  float m = -__builtin_inff();
  int i = b;
  for (; i + 3 < e; i += 4) {
    int s0 = csr[i], s1 = csr[i + 1], s2 = csr[i + 2], s3 = csr[i + 3];
    float v0 = bfu_to_f(h[s0 * FEAT + c]);
    float v1 = bfu_to_f(h[s1 * FEAT + c]);
    float v2 = bfu_to_f(h[s2 * FEAT + c]);
    float v3 = bfu_to_f(h[s3 * FEAT + c]);
    m = fmaxf(m, fmaxf(fmaxf(v0, v1), fmaxf(v2, v3)));
  }
  for (; i < e; i++)
    m = fmaxf(m, bfu_to_f(h[csr[i] * FEAT + c]));
  if (b == e) m = 0.f;                 // PyG fill for isolated nodes
  agg[n * FEAT + c] = f_to_bfu(m);
}

// ----------------- WMMA GEMM: out = relu(agg@W + b + h@R) ------------------
// Block = 128 threads = 4 waves; each wave owns a 16-row x 128-col stripe.
// Two phases (W on agg, then R on h); each phase DMAs its 32KB transposed
// weight matrix into LDS via the TDM, then the inner loop is ds_load -> wmma.
// 2 phases x 4 K-steps x 8 N-tiles = 64 v_wmma_f32_16x16x32_bf16 per wave.

__global__ void __launch_bounds__(128)
k_gemm_relu(const unsigned short* __restrict__ Abf,   // aggregated, bf16 [N][128]
            const unsigned short* __restrict__ Hbf,   // activations, bf16 [N][128]
            const unsigned short* __restrict__ Wt,    // bf16 [N=128][K=128] (W^T)
            const unsigned short* __restrict__ Rt,    // bf16 [N=128][K=128] (R^T)
            const float* __restrict__ bias,
            float* __restrict__ out32, unsigned short* __restrict__ out16,
            int ntiles, int nrows) {
  __shared__ unsigned short sB[FEAT * FEAT];          // 32 KB weight stage

  int wave   = blockIdx.x * (blockDim.x >> 5) + (threadIdx.x >> 5);
  bool active = (wave < ntiles);
  int tile   = active ? wave : (ntiles - 1);          // clamp: keep barriers legal
  int lane   = threadIdx.x & 31;
  int half   = lane >> 4;                             // K-half selector
  int l15    = lane & 15;                             // M (A) / N (B, C, D)

  // LDS byte offset of sB: low 32 bits of its flat address.
  unsigned sB_off = (unsigned)((unsigned long long)(uintptr_t)(void*)sB &
                               0xFFFFFFFFull);

  v8f acc[8] = {};                                    // 16x128 f32 accumulators

  const unsigned short* Amats[2] = {Abf, Hbf};
  const unsigned short* Bmats[2] = {Wt, Rt};
  int arow = tile * 16 + l15;
  if (arow >= nrows) arow = nrows - 1;                // address clamp only

  for (int mm = 0; mm < 2; mm++) {
    __syncthreads();                                  // prior-phase reads done
    if ((threadIdx.x >> 5) == 0) {                    // one wave issues the DMA
      tdm_load_to_lds((unsigned long long)(uintptr_t)Bmats[mm], sB_off,
                      FEAT * FEAT * 2);
      __builtin_amdgcn_s_wait_tensorcnt(0);
    }
    __syncthreads();                                  // publish LDS to block

    const unsigned short* A = Amats[mm] + (size_t)arow * FEAT;
    #pragma unroll
    for (int k0 = 0; k0 < FEAT; k0 += 32) {
      // A 16x32 in contiguous K order (matches B): lanes 0-15 hold K=k0+0..15,
      // lanes 16-31 hold K=k0+16..31 (one contiguous 32B run).
      v16bf a = *(const v16bf*)(A + k0 + half * 16);
      #pragma unroll
      for (int nt = 0; nt < 8; nt++) {
        // B 32x16: lane holds fixed N = nt*16+l15, one contiguous 32B K-run
        // (lower half K=k0..k0+15, upper half K=k0+16..k0+31) from LDS.
        v16bf bfrag =
            *(const v16bf*)(sB + (nt * 16 + l15) * FEAT + k0 + half * 16);
        acc[nt] = __builtin_amdgcn_wmma_f32_16x16x32_bf16(
            false, a, false, bfrag, (short)0, acc[nt], false, false);
      }
    }
  }

  // Epilogue: C/D layout — VGPR r holds row M = r + half*8, col N = l15.
  #pragma unroll
  for (int nt = 0; nt < 8; nt++) {
    int col  = nt * 16 + l15;
    float bv = bias[col];
    #pragma unroll
    for (int r = 0; r < 8; r++) {
      int row = tile * 16 + r + half * 8;
      if (active && row < nrows) {
        float v = acc[nt][r] + bv;
        v = v > 0.f ? v : 0.f;
        if (out32) out32[(size_t)row * FEAT + col] = v;
        else       out16[(size_t)row * FEAT + col] = f_to_bfu(v);
      }
    }
  }
}

// --------------------------------- launch ----------------------------------

extern "C" void kernel_launch(void* const* d_in, const int* in_sizes, int n_in,
                              void* d_out, int out_size, void* d_ws, size_t ws_size,
                              hipStream_t stream) {
  (void)n_in; (void)out_size; (void)ws_size;
  const float* x  = (const float*)d_in[0];
  const int*   ei = (const int*)d_in[1];
  const float* W1 = (const float*)d_in[2];
  const float* b1 = (const float*)d_in[3];
  const float* R1 = (const float*)d_in[4];
  const float* W2 = (const float*)d_in[5];
  const float* b2 = (const float*)d_in[6];
  const float* R2 = (const float*)d_in[7];

  const int N = in_sizes[0] / FEAT;    // 50000
  const int E = in_sizes[1] / 2;       // 600000
  const int* src = ei;
  const int* dst = ei + E;

  // Carve workspace (~42 MB), 256B-aligned slices.
  char* p = (char*)d_ws;
  auto take = [&](size_t bytes) {
    void* r = (void*)p;
    p += (bytes + 255) & ~(size_t)255;
    return r;
  };
  int* cnt = (int*)take((size_t)N * 4);
  int* ptr = (int*)take((size_t)(N + 1) * 4);
  int* csr = (int*)take((size_t)E * 4);
  unsigned short* hA  = (unsigned short*)take((size_t)N * FEAT * 2);
  unsigned short* hB  = (unsigned short*)take((size_t)N * FEAT * 2);
  unsigned short* agg = (unsigned short*)take((size_t)N * FEAT * 2);
  unsigned short* Wt  = (unsigned short*)take((size_t)FEAT * FEAT * 2);
  unsigned short* Rt  = (unsigned short*)take((size_t)FEAT * FEAT * 2);
  unsigned short* W2t = (unsigned short*)take((size_t)FEAT * FEAT * 2);
  unsigned short* R2t = (unsigned short*)take((size_t)FEAT * FEAT * 2);

  // CSR build (atomic fill order varies, but max-aggregation is invariant).
  k_zero_i32<<<(N + 255) / 256, 256, 0, stream>>>(cnt, N);
  k_hist<<<512, 256, 0, stream>>>(dst, cnt, E);
  k_scan<<<1, 1024, 0, stream>>>(cnt, ptr, N);
  k_fill<<<512, 256, 0, stream>>>(src, dst, ptr, cnt, csr, E);

  // bf16 activations + transposed bf16 weights.
  k_cast_bf16<<<(N * FEAT + 255) / 256, 256, 0, stream>>>(x, hA, N * FEAT);
  k_transpose_bf16<<<64, 256, 0, stream>>>(W1, Wt);
  k_transpose_bf16<<<64, 256, 0, stream>>>(R1, Rt);
  k_transpose_bf16<<<64, 256, 0, stream>>>(W2, W2t);
  k_transpose_bf16<<<64, 256, 0, stream>>>(R2, R2t);

  const int ntiles  = (N + 15) / 16;          // 3125 (exact)
  const int gblocks = (ntiles + 3) / 4;       // 4 waves per block

  // Layer 1
  k_aggregate<<<N, FEAT, 0, stream>>>(hA, ptr, csr, agg);
  k_gemm_relu<<<gblocks, 128, 0, stream>>>(agg, hA, Wt, Rt, b1,
                                           nullptr, hB, ntiles, N);
  // Layer 2
  k_aggregate<<<N, FEAT, 0, stream>>>(hB, ptr, csr, agg);
  k_gemm_relu<<<gblocks, 128, 0, stream>>>(agg, hB, W2t, R2t, b2,
                                           nullptr, hA, ntiles, N);
  // Layer 3
  k_aggregate<<<N, FEAT, 0, stream>>>(hA, ptr, csr, agg);
  k_gemm_relu<<<gblocks, 128, 0, stream>>>(agg, hA, W2t, R2t, b2,
                                           nullptr, hB, ntiles, N);
  // Layer 4 -> f32 output
  k_aggregate<<<N, FEAT, 0, stream>>>(hB, ptr, csr, agg);
  k_gemm_relu<<<gblocks, 128, 0, stream>>>(agg, hB, W2t, R2t, b2,
                                           (float*)d_out, nullptr, ntiles, N);
}
